// AttentionBlock_5677946765832
// MI455X (gfx1250) — matrix operand order repaired and checked by reading, not compiled, not run on hardware
//
#include <hip/hip_runtime.h>
#include <hip/hip_bf16.h>
#include <math.h>

typedef __bf16 bf16_t;
typedef bf16_t v16bf __attribute__((ext_vector_type(16)));
typedef bf16_t v8bf  __attribute__((ext_vector_type(8)));
typedef float  v8f   __attribute__((ext_vector_type(8)));

#define B_ 8
#define S_ 2048
#define D_ 512

union FragBF { v16bf v; v8bf h[2]; };

__device__ __forceinline__ v8f wmma_bf16(const FragBF& a, const FragBF& b, v8f c) {
  // D = A(16x32 bf16) * B(32x16 bf16) + C(16x16 f32)
  return __builtin_amdgcn_wmma_f32_16x16x32_bf16(false, a.v, false, b.v,
                                                 (short)0, c, false, false);
}

// Low 32 bits of a generic pointer to LDS == wave-relative LDS byte offset
// (flat aperture mapping).
__device__ __forceinline__ unsigned lds_off(const void* p) {
  return (unsigned)(unsigned long long)p;
}

// CDNA5 async global->LDS copy, 16B per lane, tracked by ASYNCcnt.
__device__ __forceinline__ void async_copy_b128(unsigned lds_dst, const void* gsrc) {
  asm volatile("global_load_async_to_lds_b128 %0, %1, off"
               :: "v"(lds_dst), "v"(gsrc) : "memory");
}

// ---------------------------------------------------------------------------
// Kernel 0: one-time weight transpose + bf16 convert.
// Wt[z][n][k] = (bf16) W_z[k][n].  1.5 MB result lives in the 192 MB L2 for
// the whole projection pass, so kernel 1 can read B-fragments straight from
// global with contiguous 16-B loads (no LDS transpose).
// ---------------------------------------------------------------------------
__global__ __launch_bounds__(256)
void wt_kernel(const float* __restrict__ Wq, const float* __restrict__ Wk,
               const float* __restrict__ Wv, bf16_t* __restrict__ Wt) {
  const int idx = blockIdx.x * 256 + threadIdx.x;       // [0, 3*D*D)
  const int z   = idx / (D_ * D_);
  const int rem = idx % (D_ * D_);
  const int k   = rem / D_;
  const int n   = rem % D_;                             // coalesced read along n
  const float* W = (z == 0) ? Wq : (z == 1) ? Wk : Wv;
  Wt[(size_t)z * D_ * D_ + (size_t)n * D_ + k] = (bf16_t)W[(size_t)k * D_ + n];
}

// ---------------------------------------------------------------------------
// Kernel 1: Q/K/V projections (bf16 WMMA, fp32 accum).  LDS-free: A-fragments
// are packed from global x (each x row feeds exactly one wave), B-fragments
// are contiguous reads of the pre-transposed L2-resident Wt.
// Q pre-scaled by 1/sqrt(D).  V written TRANSPOSED [B][D][S] so the attention
// kernel can stream V tiles with contiguous async B128 copies.
// ---------------------------------------------------------------------------
__global__ __launch_bounds__(128)
void qkv_proj_kernel(const float* __restrict__ x,
                     const bf16_t* __restrict__ Wt,
                     bf16_t* __restrict__ wsQ,
                     bf16_t* __restrict__ wsK,
                     bf16_t* __restrict__ wsVt) {
  const int tid  = threadIdx.x;
  const int wave = tid >> 5, lane = tid & 31;
  const int half = lane >> 4, lr = lane & 15;
  const int m0 = blockIdx.x * 64;                       // global row base (B*S rows)
  const int n0 = blockIdx.y * 64;                       // output col base
  const int which = blockIdx.z;                         // 0=Q 1=K 2=V
  const bf16_t* Wtz = Wt + (size_t)which * D_ * D_;
  const float oscale = (which == 0) ? 0.044194173824159216f : 1.0f;  // 1/sqrt(512)

  const float* xrow = x + (size_t)(m0 + wave * 16 + lr) * D_;

  v8f acc[4] = {{}, {}, {}, {}};

  for (int kt = 0; kt < D_ / 32; ++kt) {
    const int k0 = kt * 32;
    // A fragment straight from global fp32 x: chunks k0+half*16 and k0+half*16+8
    FragBF af;
    {
      const float* c0 = xrow + k0 + half * 16;
      const float* c1 = xrow + k0 + half * 16 + 8;
      float4 f0 = ((const float4*)c0)[0];
      float4 f1 = ((const float4*)c0)[1];
      float4 f2 = ((const float4*)c1)[0];
      float4 f3 = ((const float4*)c1)[1];
      af.v[0]  = (bf16_t)f0.x; af.v[1]  = (bf16_t)f0.y;
      af.v[2]  = (bf16_t)f0.z; af.v[3]  = (bf16_t)f0.w;
      af.v[4]  = (bf16_t)f1.x; af.v[5]  = (bf16_t)f1.y;
      af.v[6]  = (bf16_t)f1.z; af.v[7]  = (bf16_t)f1.w;
      af.v[8]  = (bf16_t)f2.x; af.v[9]  = (bf16_t)f2.y;
      af.v[10] = (bf16_t)f2.z; af.v[11] = (bf16_t)f2.w;
      af.v[12] = (bf16_t)f3.x; af.v[13] = (bf16_t)f3.y;
      af.v[14] = (bf16_t)f3.z; af.v[15] = (bf16_t)f3.w;
    }
    // B fragments: contiguous bf16 reads of Wt rows (L2-resident)
    FragBF bfr[4];
    #pragma unroll
    for (int j = 0; j < 4; ++j) {
      const bf16_t* bp = Wtz + (size_t)(n0 + j * 16 + lr) * D_ + k0 + half * 16;
      bfr[j].h[0] = *(const v8bf*)&bp[0];
      bfr[j].h[1] = *(const v8bf*)&bp[8];
    }
    #pragma unroll
    for (int j = 0; j < 4; ++j)
      acc[j] = wmma_bf16(af, bfr[j], acc[j]);
  }
  // Epilogue: C layout lane -> (m = half*8+r, n = lr)
  #pragma unroll
  for (int j = 0; j < 4; ++j) {
    #pragma unroll
    for (int r = 0; r < 8; ++r) {
      const int row = m0 + wave * 16 + half * 8 + r;    // row in [0, B*S)
      const int col = n0 + j * 16 + lr;
      const bf16_t val = (bf16_t)(acc[j][r] * oscale);
      if (which == 2) {
        // V stored transposed: [B][D][S]
        const int bb = row >> 11;                       // row / S_
        const int ss = row & (S_ - 1);
        wsVt[((size_t)bb * D_ + col) * S_ + ss] = val;
      } else {
        bf16_t* dst = (which == 0) ? wsQ : wsK;
        dst[(size_t)row * D_ + col] = val;
      }
    }
  }
}

// ---------------------------------------------------------------------------
// Kernel 2: flash attention with double-buffered async K/V tiles.
// One WG = (batch, 16 query rows); 4 waves redundantly compute the 16x32
// score tile, each wave owns a 128-dim slice of the P*V output.
// K/V tiles for step kt+1 are prefetched with GLOBAL_LOAD_ASYNC_TO_LDS_B128
// while step kt runs on the WMMA pipe.
// ---------------------------------------------------------------------------
#define KSTR 520    // K tile stride  [32 keys][520]   (16640 bf16 / buf)
#define VSTR 40     // V^T tile stride [512 dims][40]  (20480 bf16 / buf)
#define PSTR 40
#define NT   (S_ / 32)

__global__ __launch_bounds__(128)
void attn_kernel(const bf16_t* __restrict__ Q,
                 const bf16_t* __restrict__ K,
                 const bf16_t* __restrict__ Vt,
                 float* __restrict__ out) {
  __shared__ __align__(16) bf16_t ldsK [2][32 * KSTR];    // 2 x 33.3 KB
  __shared__ __align__(16) bf16_t ldsVt[2][512 * VSTR];   // 2 x 40 KB
  __shared__ __align__(16) bf16_t ldsP [4 * 16 * PSTR];   // 5 KB

  const int tid  = threadIdx.x;
  const int wave = tid >> 5, lane = tid & 31;
  const int half = lane >> 4, lr = lane & 15;
  const int b  = blockIdx.y;
  const int q0 = blockIdx.x * 16;

  // Per-thread async-copy source/dest bases (16 B128 copies each for K and Vt)
  const int krow   = tid >> 2;            // 0..31
  const int kdbase = (tid & 3) * 128;     // 128 bf16 along d
  const int vrow0  = tid * 4;             // 4 dim-rows per thread

  // Q A-fragments for all 16 k-steps straight from global (row-major bf16
  // matches the A-frag chunk layout: two 16-byte runs per lane).
  FragBF qf[16];
  {
    const bf16_t* qrow = Q + (size_t)(b * S_ + q0 + lr) * D_;
    #pragma unroll
    for (int kk = 0; kk < 16; ++kk) {
      qf[kk].h[0] = *(const v8bf*)&qrow[kk * 32 + half * 16];
      qf[kk].h[1] = *(const v8bf*)&qrow[kk * 32 + half * 16 + 8];
    }
  }

  v8f o[8] = {{}, {}, {}, {}, {}, {}, {}, {}};
  float mrow[8], lrow[8];
  #pragma unroll
  for (int r = 0; r < 8; ++r) { mrow[r] = -1e30f; lrow[r] = 0.0f; }

  bf16_t* myP = &ldsP[wave * 16 * PSTR];

  // Issue a tile prefetch (32 async B128 ops per wave: 16 K + 16 Vt).
  auto prefetch = [&](int kt, int buf) {
    const int kbase = kt * 32;
    {
      const bf16_t* src = K + (size_t)(b * S_ + kbase + krow) * D_ + kdbase;
      unsigned dst = lds_off(&ldsK[buf][krow * KSTR + kdbase]);
      #pragma unroll
      for (int i = 0; i < 16; ++i)
        async_copy_b128(dst + i * 16, src + i * 8);
    }
    {
      #pragma unroll
      for (int rr = 0; rr < 4; ++rr) {
        const int d = vrow0 + rr;
        const bf16_t* src = Vt + ((size_t)b * D_ + d) * S_ + kbase;
        unsigned dst = lds_off(&ldsVt[buf][d * VSTR]);
        #pragma unroll
        for (int i = 0; i < 4; ++i)
          async_copy_b128(dst + i * 16, src + i * 8);
      }
    }
  };

  prefetch(0, 0);

  for (int kt = 0; kt < NT; ++kt) {
    const int cur = kt & 1;
    __syncthreads();                       // all waves done reading buf[cur^1]
    if (kt + 1 < NT) {
      prefetch(kt + 1, cur ^ 1);
      asm volatile("s_wait_asynccnt 32" ::: "memory");   // tile kt landed (mine)
    } else {
      asm volatile("s_wait_asynccnt 0" ::: "memory");
    }
    __syncthreads();                       // tile kt landed (all waves)

    const bf16_t* kb = &ldsK[cur][0];
    const bf16_t* vb = &ldsVt[cur][0];

    // Scores S[16 q][32 k]: pipelined B-frag loads + WMMAs
    v8f sc0 = {}, sc1 = {};
    {
      FragBF b0, b1, p0f, p1f;
      const bf16_t* c0 = &kb[lr * KSTR + half * 16];
      const bf16_t* c1 = &kb[(16 + lr) * KSTR + half * 16];
      b0.h[0] = *(const v8bf*)&c0[0];  b0.h[1] = *(const v8bf*)&c0[8];
      b1.h[0] = *(const v8bf*)&c1[0];  b1.h[1] = *(const v8bf*)&c1[8];
      #pragma unroll
      for (int kk = 0; kk < 16; ++kk) {
        if (kk < 15) {
          const bf16_t* n0 = &kb[lr * KSTR + (kk + 1) * 32 + half * 16];
          const bf16_t* n1 = &kb[(16 + lr) * KSTR + (kk + 1) * 32 + half * 16];
          p0f.h[0] = *(const v8bf*)&n0[0];  p0f.h[1] = *(const v8bf*)&n0[8];
          p1f.h[0] = *(const v8bf*)&n1[0];  p1f.h[1] = *(const v8bf*)&n1[8];
        }
        sc0 = wmma_bf16(qf[kk], b0, sc0);
        sc1 = wmma_bf16(qf[kk], b1, sc1);
        b0 = p0f; b1 = p1f;
      }
    }

    // Online softmax (registers + per-wave ldsP only)
    FragBF pf;
    #pragma unroll
    for (int r = 0; r < 8; ++r) {
      float s = fmaxf(sc0[r], sc1[r]);
      s = fmaxf(s, __shfl_xor(s, 1));
      s = fmaxf(s, __shfl_xor(s, 2));
      s = fmaxf(s, __shfl_xor(s, 4));
      s = fmaxf(s, __shfl_xor(s, 8));      // row max (stays within 16-lane half)
      const float mnew  = fmaxf(mrow[r], s);
      const float alpha = __expf(mrow[r] - mnew);
      const float e0 = __expf(sc0[r] - mnew);
      const float e1 = __expf(sc1[r] - mnew);
      float rs = e0 + e1;
      rs += __shfl_xor(rs, 1); rs += __shfl_xor(rs, 2);
      rs += __shfl_xor(rs, 4); rs += __shfl_xor(rs, 8);
      lrow[r] = lrow[r] * alpha + rs;
      mrow[r] = mnew;
      #pragma unroll
      for (int j = 0; j < 8; ++j) o[j][r] *= alpha;
      // P in C-layout (m = half*8+r, n = lr / 16+lr) -> per-wave LDS scratch
      myP[(half * 8 + r) * PSTR + lr]      = (bf16_t)e0;
      myP[(half * 8 + r) * PSTR + 16 + lr] = (bf16_t)e1;
    }
    // Re-read P as an A-fragment (same-wave DS ops are ordered)
    {
      const bf16_t* prow = &myP[lr * PSTR];
      pf.h[0] = *(const v8bf*)&prow[half * 16];
      pf.h[1] = *(const v8bf*)&prow[half * 16 + 8];
    }

    // O += P(16x32) * V(32 x 128-slice): pipelined B-frag loads + 8 WMMAs
    {
      FragBF vf, nvf;
      const bf16_t* v0 = &vb[(wave * 128 + lr) * VSTR + half * 16];
      vf.h[0] = *(const v8bf*)&v0[0];  vf.h[1] = *(const v8bf*)&v0[8];
      #pragma unroll
      for (int j = 0; j < 8; ++j) {
        if (j < 7) {
          const bf16_t* vn = &vb[(wave * 128 + (j + 1) * 16 + lr) * VSTR + half * 16];
          nvf.h[0] = *(const v8bf*)&vn[0];  nvf.h[1] = *(const v8bf*)&vn[8];
        }
        o[j] = wmma_bf16(pf, vf, o[j]);
        vf = nvf;
      }
    }
  }

  // Epilogue: out = O / l  (hoisted reciprocals)
  float linv[8];
  #pragma unroll
  for (int r = 0; r < 8; ++r) linv[r] = 1.0f / lrow[r];
  #pragma unroll
  for (int j = 0; j < 8; ++j) {
    #pragma unroll
    for (int r = 0; r < 8; ++r) {
      const int row = q0 + half * 8 + r;
      const int col = wave * 128 + j * 16 + lr;
      out[(size_t)(b * S_ + row) * D_ + col] = o[j][r] * linv[r];
    }
  }
}

// ---------------------------------------------------------------------------
extern "C" void kernel_launch(void* const* d_in, const int* in_sizes, int n_in,
                              void* d_out, int out_size, void* d_ws, size_t ws_size,
                              hipStream_t stream) {
  const float* x  = (const float*)d_in[0];
  const float* Wq = (const float*)d_in[1];
  const float* Wk = (const float*)d_in[2];
  const float* Wv = (const float*)d_in[3];
  float* out = (float*)d_out;

  const size_t elems = (size_t)B_ * S_ * D_;
  bf16_t* wsQ  = (bf16_t*)d_ws;         // 16 MB
  bf16_t* wsK  = wsQ + elems;           // 16 MB
  bf16_t* wsVt = wsK + elems;           // 16 MB, layout [B][D][S]
  bf16_t* wsWt = wsVt + elems;          // 1.5 MB, layout [3][D][D] (n-major)

  wt_kernel<<<3 * D_ * D_ / 256, 256, 0, stream>>>(Wq, Wk, Wv, wsWt);

  dim3 g1((B_ * S_) / 64, D_ / 64, 3);  // 256 x 8 x 3 workgroups
  qkv_proj_kernel<<<g1, 128, 0, stream>>>(x, wsWt, wsQ, wsK, wsVt);

  dim3 g2(S_ / 16, B_);                 // 128 x 8 workgroups
  attn_kernel<<<g2, 128, 0, stream>>>(wsQ, wsK, wsVt, out);
}
